// BPUMultiHeadedAttention_34110630265179
// MI455X (gfx1250) — matrix operand order repaired and checked by reading, not compiled, not run on hardware
//
#include <hip/hip_runtime.h>

// ---------------------------------------------------------------------------
// BPU MultiHeadedAttention for MI455X (gfx1250, wave32, WMMA bf16 path)
//
// H=16 heads, DK=64, CHUNK=512, CACHE_T=7680, TIME=8192, NFEAT=1024
//
// Roofline: ~21.5 GFLOP vs ~210 MB of HBM traffic -> memory-bound at bf16
// matrix rate; all GEMMs/attention go through v_wmma_f32_16x16x32_bf16 with
// f32 accumulation, tiles staged via global_load_async_to_lds (ASYNCcnt).
// ---------------------------------------------------------------------------

#define H      16
#define DK     64
#define CHUNK  512
#define CACHET 7680
#define TIME   8192
#define NFEAT  1024

typedef __attribute__((ext_vector_type(16))) __bf16 v16bf;
typedef __attribute__((ext_vector_type(8)))  float  v8f;

union Frag16 { unsigned u[8]; v16bf v; };

__device__ __forceinline__ unsigned short f32_bf16(float f) {
  union { float f; unsigned u; } a; a.f = f;
  unsigned u = a.u;
  unsigned r = u + 0x7FFFu + ((u >> 16) & 1u);   // round-to-nearest-even
  return (unsigned short)(r >> 16);
}

__device__ __forceinline__ unsigned pack_bf16x2(float f0, float f1) {
#if __has_builtin(__builtin_amdgcn_cvt_pk_bf16_f32)
  typedef __attribute__((ext_vector_type(2))) __bf16 v2bf;
  union { v2bf v; unsigned u; } c;
  c.v = __builtin_amdgcn_cvt_pk_bf16_f32(f0, f1);
  return c.u;
#else
  return (unsigned)f32_bf16(f0) | ((unsigned)f32_bf16(f1) << 16);
#endif
}

// Low 32 bits of a generic (flat) pointer to a __shared__ object are the LDS
// byte offset (ISA 10.2: LDS aperture uses addr[31:0] relative to wave LDS).
__device__ __forceinline__ unsigned lds_off(const void* p) {
  return (unsigned)(unsigned long long)p;
}

// CDNA5 async copy global->LDS, tracked on ASYNCcnt (CDNA5 ISA)
__device__ __forceinline__ void async_load_b32(const void* gaddr, unsigned ldsoff) {
  asm volatile("global_load_async_to_lds_b32 %0, %1, off"
               :: "v"(ldsoff), "v"((unsigned long long)gaddr)
               : "memory");
}
__device__ __forceinline__ void wait_async0() {
  asm volatile("s_wait_asynccnt 0" ::: "memory");
}

// ---------------------------------------------------------------------------
// Generic GEMM  Y[m,n] = sum_k W[m,k] * X[k,n] + b[m]
// One wave computes one 16x16 tile; K consumed 32/step. f32 tiles are staged
// with async-to-LDS copies (B transposed for free via per-lane LDS dests),
// converted to bf16 while gathering WMMA fragments.
// ---------------------------------------------------------------------------
__global__ __launch_bounds__(32)
void gemm_wmma(const float* __restrict__ W, const float* __restrict__ X,
               const float* __restrict__ bias, float* __restrict__ Y,
               int M, int N, int K) {
  __shared__ float ldsAf[16 * 32];                  // [m][k] f32
  __shared__ float ldsBf[16 * 32];                  // [n][k] f32 (transposed)
  const int lane = threadIdx.x;
  const int hw   = lane >> 4;                       // half-wave (0/1)
  const int l16  = lane & 15;
  const int n0 = blockIdx.x * 16;
  const int m0 = blockIdx.y * 16;

  v8f acc = {};
  for (int k0 = 0; k0 < K; k0 += 32) {
    if (k0 + 32 < K) {                              // global_prefetch_b8
      __builtin_prefetch(&W[(size_t)m0 * K + k0 + 32], 0, 0);
      __builtin_prefetch(&X[(size_t)(k0 + 32) * N + n0], 0, 0);
    }
    // A tile 16x32 (coalesced rows), async global->LDS
    #pragma unroll
    for (int i = 0; i < 16; ++i)
      async_load_b32(&W[(size_t)(m0 + i) * K + k0 + lane],
                     lds_off(&ldsAf[i * 32 + lane]));
    // B tile 32x16 -> LDS transposed as [n][k] (scatter on the LDS side)
    #pragma unroll
    for (int i = 0; i < 16; ++i) {
      int r = i * 2 + hw;
      async_load_b32(&X[(size_t)(k0 + r) * N + n0 + l16],
                     lds_off(&ldsBf[l16 * 32 + r]));
    }
    wait_async0();
    __syncthreads();

    // Gather WMMA fragments in contiguous k order (k = 16*hw + s, same for A and B), cvt f32->bf16
    Frag16 fa, fb;
    #pragma unroll
    for (int j = 0; j < 8; ++j) {
      int ka = 2 * j + (hw ? 16 : 0);
      const float* pa = &ldsAf[l16 * 32 + ka];
      fa.u[j] = pack_bf16x2(pa[0], pa[1]);
      int kb = 2 * j + (hw ? 16 : 0);
      const float* pb = &ldsBf[l16 * 32 + kb];
      fb.u[j] = pack_bf16x2(pb[0], pb[1]);
    }
    acc = __builtin_amdgcn_wmma_f32_16x16x32_bf16(false, fa.v, false, fb.v,
                                                  (short)0, acc, false, false);
    __syncthreads();
  }
  #pragma unroll
  for (int r = 0; r < 8; ++r) {
    int m = r + (hw ? 8 : 0);
    Y[(size_t)(m0 + m) * N + n0 + l16] = acc[r] + bias[m0 + m];
  }
}

// ---------------------------------------------------------------------------
// new_cache = concat(k_full, v_full) : (H, 128, TIME), written to d_out tail
// ---------------------------------------------------------------------------
__global__ void concat_cache(const float* __restrict__ cache,
                             const float* __restrict__ kproj,
                             const float* __restrict__ vproj,
                             float* __restrict__ ncache) {
  size_t idx = (size_t)blockIdx.x * blockDim.x + threadIdx.x;
  if (idx >= (size_t)H * 128 * TIME) return;
  int t = (int)(idx % TIME);
  int r = (int)((idx / TIME) % 128);
  int h = (int)(idx / ((size_t)TIME * 128));
  float v;
  if (t < CACHET) {
    v = cache[((size_t)h * 128 + r) * CACHET + t];
  } else {
    int d = r & 63;
    const float* src = (r < 64) ? kproj : vproj;
    v = src[((size_t)h * DK + d) * CHUNK + (t - CACHET)];
  }
  ncache[idx] = v;
}

// ---------------------------------------------------------------------------
// Build bf16 attention operands:
//   KbfT[h][t][d]  (pairs contiguous in d -> scores B-fragments)
//   Vbf [h][d][t]  (pairs contiguous in t -> AV B-fragments)
// ---------------------------------------------------------------------------
__global__ void build_kv_bf16(const float* __restrict__ cache,
                              const float* __restrict__ kproj,
                              const float* __restrict__ vproj,
                              unsigned short* __restrict__ KbfT,
                              unsigned short* __restrict__ Vbf) {
  size_t idx = (size_t)blockIdx.x * blockDim.x + threadIdx.x;
  if (idx >= (size_t)H * TIME * DK) return;
  int d = (int)(idx % DK);
  int t = (int)((idx / DK) % TIME);
  int h = (int)(idx / ((size_t)DK * TIME));
  float kv, vv;
  if (t < CACHET) {
    kv = cache[((size_t)h * 128 + d) * CACHET + t];
    vv = cache[((size_t)h * 128 + 64 + d) * CACHET + t];
  } else {
    kv = kproj[((size_t)h * DK + d) * CHUNK + (t - CACHET)];
    vv = vproj[((size_t)h * DK + d) * CHUNK + (t - CACHET)];
  }
  KbfT[((size_t)h * TIME + t) * DK + d] = f32_bf16(kv);
  Vbf[((size_t)h * DK + d) * TIME + t] = f32_bf16(vv);
}

// qbf[h][c][d] = q_proj[h*DK+d][c] * rsqrt(DK)   (pre-scaled, transposed)
__global__ void build_q_bf16(const float* __restrict__ qproj,
                             unsigned short* __restrict__ qbf) {
  size_t idx = (size_t)blockIdx.x * blockDim.x + threadIdx.x;
  if (idx >= (size_t)H * CHUNK * DK) return;
  int d = (int)(idx % DK);
  int c = (int)((idx / DK) % CHUNK);
  int h = (int)(idx / ((size_t)DK * CHUNK));
  qbf[idx] = f32_bf16(qproj[((size_t)h * DK + d) * CHUNK + c] * 0.125f);
}

// ---------------------------------------------------------------------------
// Flash attention: one wave per (head, 16 q-rows). Streams TIME in 32-steps.
//   S = qT x K  (4 wmma), online softmax (shfl_xor row reductions),
//   P -> LDS -> A-fragment, acc += P x V^T (4 wmma). Output x[h][d][c].
// Mask tiles staged with async-to-LDS copies.
// ---------------------------------------------------------------------------
__global__ __launch_bounds__(32)
void attn_wmma(const unsigned short* __restrict__ qbf,
               const unsigned short* __restrict__ KbfT,
               const unsigned short* __restrict__ Vbf,
               const unsigned char* __restrict__ mask,
               float* __restrict__ xatt) {
  __shared__ unsigned short ldsP[16 * 32];          // P tile bf16 [c][t]
  __shared__ unsigned char  ldsM[16 * 32];          // mask tile  [c][t]
  const int lane = threadIdx.x;
  const int hw   = lane >> 4;
  const int l16  = lane & 15;
  const int h  = blockIdx.y;
  const int c0 = blockIdx.x * 16;
  const float NEG_INF = -__builtin_inff();

  // q A-fragments for K-halves 0..31 and 32..63 (stay resident)
  Frag16 aq[2];
  #pragma unroll
  for (int kh = 0; kh < 2; ++kh)
    #pragma unroll
    for (int j = 0; j < 8; ++j) {
      int ka = (2 * j + (hw ? 16 : 0)) + kh * 32;
      aq[kh].u[j] = *(const unsigned*)&qbf[((size_t)h * CHUNK + c0 + l16) * DK + ka];
    }

  v8f acc0 = {}, acc1 = {}, acc2 = {}, acc3 = {};
  float rmax[8], rsum[8];
  #pragma unroll
  for (int r = 0; r < 8; ++r) { rmax[r] = NEG_INF; rsum[r] = 0.f; }

  for (int t0 = 0; t0 < TIME; t0 += 32) {
    // ---- stage 16x32 mask tile via async global->LDS ----
    const unsigned char* mbase = mask + ((size_t)h * CHUNK + c0) * TIME + t0;
    #pragma unroll
    for (int i = 0; i < 4; ++i) {
      int u = i * 32 + lane;
      int row = u >> 3, col = (u & 7) * 4;
      async_load_b32(&mbase[(size_t)row * TIME + col],
                     lds_off(&ldsM[row * 32 + col]));
    }
    wait_async0();
    __syncthreads();

    // ---- scores: two 16x16 N-tiles, each K=64 via two wmma ----
    v8f s[2];
    #pragma unroll
    for (int nt = 0; nt < 2; ++nt) {
      v8f z = {};
      #pragma unroll
      for (int kh = 0; kh < 2; ++kh) {
        Frag16 bk;
        int t = t0 + nt * 16 + l16;
        #pragma unroll
        for (int j = 0; j < 8; ++j) {
          int dk = kh * 32 + 2 * j + (hw ? 16 : 0);
          bk.u[j] = *(const unsigned*)&KbfT[((size_t)h * TIME + t) * DK + dk];
        }
        z = __builtin_amdgcn_wmma_f32_16x16x32_bf16(false, aq[kh].v, false, bk.v,
                                                    (short)0, z, false, false);
      }
      s[nt] = z;
    }

    // ---- online softmax (rows live across one half-wave lane group) ----
    #pragma unroll
    for (int r = 0; r < 8; ++r) {
      int mrow = r + hw * 8;
      float v0 = ldsM[mrow * 32 + l16]      ? s[0][r] : NEG_INF;
      float v1 = ldsM[mrow * 32 + 16 + l16] ? s[1][r] : NEG_INF;
      float m = fmaxf(v0, v1);
      m = fmaxf(m, __shfl_xor(m, 1, 16));
      m = fmaxf(m, __shfl_xor(m, 2, 16));
      m = fmaxf(m, __shfl_xor(m, 4, 16));
      m = fmaxf(m, __shfl_xor(m, 8, 16));
      float mx = fmaxf(rmax[r], m);
      float scale = (mx > NEG_INF) ? ((rmax[r] > NEG_INF) ? __expf(rmax[r] - mx) : 0.f)
                                   : 0.f;
      float p0 = (v0 > NEG_INF) ? __expf(v0 - mx) : 0.f;
      float p1 = (v1 > NEG_INF) ? __expf(v1 - mx) : 0.f;
      float ps = p0 + p1;
      ps += __shfl_xor(ps, 1, 16);
      ps += __shfl_xor(ps, 2, 16);
      ps += __shfl_xor(ps, 4, 16);
      ps += __shfl_xor(ps, 8, 16);
      rsum[r] = rsum[r] * scale + ps;
      rmax[r] = mx;
      acc0[r] *= scale; acc1[r] *= scale; acc2[r] *= scale; acc3[r] *= scale;
      ldsP[mrow * 32 + l16]      = f32_bf16(p0);
      ldsP[mrow * 32 + 16 + l16] = f32_bf16(p1);
    }
    __syncthreads();

    // ---- acc[c,d] += P(16x32) x V^T(32x16) for 4 d-groups ----
    Frag16 ap;
    #pragma unroll
    for (int j = 0; j < 8; ++j) {
      int ka = 2 * j + (hw ? 16 : 0);
      ap.u[j] = *(const unsigned*)&ldsP[l16 * 32 + ka];
    }
    #pragma unroll
    for (int g = 0; g < 4; ++g) {
      Frag16 bv;
      int d = g * 16 + l16;
      #pragma unroll
      for (int j = 0; j < 8; ++j) {
        int tt = t0 + 2 * j + (hw ? 16 : 0);
        bv.u[j] = *(const unsigned*)&Vbf[((size_t)h * DK + d) * TIME + tt];
      }
      v8f* accp = (g == 0) ? &acc0 : (g == 1) ? &acc1 : (g == 2) ? &acc2 : &acc3;
      *accp = __builtin_amdgcn_wmma_f32_16x16x32_bf16(false, ap.v, false, bv.v,
                                                      (short)0, *accp, false, false);
    }
    __syncthreads();
  }

  // ---- normalize + write x[h*DK+d][c] ----
  #pragma unroll
  for (int g = 0; g < 4; ++g) {
    const v8f* accp = (g == 0) ? &acc0 : (g == 1) ? &acc1 : (g == 2) ? &acc2 : &acc3;
    int d = g * 16 + l16;
    #pragma unroll
    for (int r = 0; r < 8; ++r) {
      int c = c0 + r + hw * 8;
      xatt[((size_t)h * DK + d) * CHUNK + c] = (*accp)[r] / rsum[r];
    }
  }
}

// ---------------------------------------------------------------------------
extern "C" void kernel_launch(void* const* d_in, const int* in_sizes, int n_in,
                              void* d_out, int out_size, void* d_ws, size_t ws_size,
                              hipStream_t stream) {
  const float* q     = (const float*)d_in[0];
  const float* k     = (const float*)d_in[1];
  const float* v     = (const float*)d_in[2];
  const unsigned char* mask = (const unsigned char*)d_in[3];
  const float* cache = (const float*)d_in[4];
  const float* Wq = (const float*)d_in[5];  const float* bq = (const float*)d_in[6];
  const float* Wk = (const float*)d_in[7];  const float* bk = (const float*)d_in[8];
  const float* Wv = (const float*)d_in[9];  const float* bv = (const float*)d_in[10];
  const float* Wo = (const float*)d_in[11]; const float* bo = (const float*)d_in[12];

  float* out      = (float*)d_out;                       // x_out: 1024*512
  float* ncache   = out + (size_t)NFEAT * CHUNK;         // new_cache: 16*128*8192

  char* ws = (char*)d_ws;
  float* qproj = (float*)(ws + 0);                       // 2 MB
  float* kproj = (float*)(ws + (2u << 20));              // 2 MB
  float* vproj = (float*)(ws + (4u << 20));              // 2 MB
  float* xatt  = (float*)(ws + (6u << 20));              // 2 MB
  unsigned short* qbf  = (unsigned short*)(ws + (8u << 20));   // 1 MB
  unsigned short* KbfT = (unsigned short*)(ws + (9u << 20));   // 16 MB
  unsigned short* Vbf  = (unsigned short*)(ws + (25u << 20));  // 16 MB

  dim3 ggrid(CHUNK / 16, NFEAT / 16);   // 32 x 64 tiles, 1 wave each
  gemm_wmma<<<ggrid, 32, 0, stream>>>(Wq, q, bq, qproj, NFEAT, CHUNK, NFEAT);
  gemm_wmma<<<ggrid, 32, 0, stream>>>(Wk, k, bk, kproj, NFEAT, CHUNK, NFEAT);
  gemm_wmma<<<ggrid, 32, 0, stream>>>(Wv, v, bv, vproj, NFEAT, CHUNK, NFEAT);

  {
    size_t n = (size_t)H * 128 * TIME;
    concat_cache<<<(unsigned)((n + 255) / 256), 256, 0, stream>>>(cache, kproj, vproj, ncache);
  }
  {
    size_t n = (size_t)H * TIME * DK;
    build_kv_bf16<<<(unsigned)((n + 255) / 256), 256, 0, stream>>>(cache, kproj, vproj, KbfT, Vbf);
  }
  {
    size_t n = (size_t)H * CHUNK * DK;
    build_q_bf16<<<(unsigned)((n + 255) / 256), 256, 0, stream>>>(qproj, qbf);
  }

  dim3 agrid(CHUNK / 16, H);            // 32 q-tiles x 16 heads
  attn_wmma<<<agrid, 32, 0, stream>>>(qbf, KbfT, Vbf, mask, xatt);

  gemm_wmma<<<ggrid, 32, 0, stream>>>(Wo, xatt, bo, out, NFEAT, CHUNK, NFEAT);
}
